// MultiHeadAttention_13572096655786
// MI455X (gfx1250) — matrix operand order repaired and checked by reading, not compiled, not run on hardware
//
#include <hip/hip_runtime.h>
#include <math.h>

typedef __bf16 bf16;
typedef __attribute__((ext_vector_type(4)))  __bf16 bf16x4;
typedef __attribute__((ext_vector_type(8)))  __bf16 bf16x8;
typedef __attribute__((ext_vector_type(16))) __bf16 v16bf;
typedef __attribute__((ext_vector_type(8)))  float  v8f;

#define BLK_M 128
#define BLK_N 128
#define BLK_K 32
#define SA 40          // padded LDS row stride (elems) -> 80B; balanced 2-cycle b128 access
#define SCALE 0.022097086912079608f   // 2048^-0.5 (reference scales by n_embed^-0.5)

// ---------------- WMMA helpers (CDNA5 wave32, 16x16x32 bf16 -> f32) ----------------

__device__ __forceinline__ v8f wmma_bf16(v16bf a, v16bf b, v8f c) {
  return __builtin_amdgcn_wmma_f32_16x16x32_bf16(false, a, false, b, (short)0, c, false, false);
}

// A fragment (16x32): lane L row = row0+(L&15); elem e K = ((L>=16)?8:0)+(e<8?e:e+8)
__device__ __forceinline__ v16bf load_a_frag(const bf16* As, int row0) {
  const int lane = threadIdx.x & 31;
  const bf16* p = As + (row0 + (lane & 15)) * SA;
  const int hi = (lane >> 4) << 3;
  bf16x8 c0 = *(const bf16x8*)(p + hi);
  bf16x8 c1 = *(const bf16x8*)(p + hi + 16);
  v16bf a;
#pragma unroll
  for (int i = 0; i < 8; i++) { a[i] = c0[i]; a[i + 8] = c1[i]; }
  return a;
}

// B fragment (32x16, LDS stored N-major): lane L col = col0+(L&15); elem e K = ((L>=16)?8:0)+(e<8?e:e+8)
__device__ __forceinline__ v16bf load_b_frag(const bf16* Bs, int col0) {
  const int lane = threadIdx.x & 31;
  const bf16* p = Bs + (col0 + (lane & 15)) * SA + ((lane >> 4) << 3);
  bf16x8 c0 = *(const bf16x8*)(p);
  bf16x8 c1 = *(const bf16x8*)(p + 16);
  v16bf b;
#pragma unroll
  for (int i = 0; i < 8; i++) { b[i] = c0[i]; b[i + 8] = c1[i]; }
  return b;
}

// Per-K-step compute: 2 A frags x 4 B frags -> 8 WMMAs
__device__ __forceinline__ void mma_step(const bf16* As, const bf16* Bs,
                                         int wm, int wn, v8f acc[2][4]) {
  v16bf a0 = load_a_frag(As, wm);
  v16bf a1 = load_a_frag(As, wm + 16);
#pragma unroll
  for (int ni = 0; ni < 4; ni++) {
    v16bf b = load_b_frag(Bs, wn + ni * 16);
    acc[0][ni] = wmma_bf16(a0, b, acc[0][ni]);
    acc[1][ni] = wmma_bf16(a1, b, acc[1][ni]);
  }
}

// ---------------- Kernel 1: fused QKV projection ----------------
// out[m,n]=sum_k x[m,k]*W[n,k]+bias[n]; M=32768(b,t), N=6144(q|k|v), K=2048
// q,k stored (B,H,T,D); v stored transposed (B,H,D,T).
__global__ __launch_bounds__(256)
void qkv_gemm(const float* __restrict__ x,
              const float* __restrict__ Wq, const float* __restrict__ Wk, const float* __restrict__ Wv,
              const float* __restrict__ bq, const float* __restrict__ bk, const float* __restrict__ bv,
              bf16* __restrict__ qws, bf16* __restrict__ kws, bf16* __restrict__ vtws) {
  __shared__ bf16 As[2][BLK_M * SA];
  __shared__ bf16 Bs[2][BLK_N * SA];
  const int tid = threadIdx.x, lane = tid & 31, wave = tid >> 5;
  const int wm = (wave & 3) * 32, wn = (wave >> 2) * 64;
  const int m0 = blockIdx.y * BLK_M, n0 = blockIdx.x * BLK_N;
  const int which = n0 >> 11;                       // BLK_N divides 2048
  const float* W    = (which == 0) ? Wq : (which == 1) ? Wk : Wv;
  const float* bias = (which == 0) ? bq : (which == 1) ? bk : bv;
  const int nr0 = n0 & 2047;
  const int rA = (tid) >> 3, cA = (tid & 7) << 2;   // per-i: row = rA + i*32

  float4 ra[4], rb[4];
  v8f acc[2][4] = {};

  auto load_g = [&](int k0) {
#pragma unroll
    for (int i = 0; i < 4; i++)
      ra[i] = *(const float4*)(x + (size_t)(m0 + rA + i * 32) * 2048 + k0 + cA);
#pragma unroll
    for (int i = 0; i < 4; i++)
      rb[i] = *(const float4*)(W + (size_t)(nr0 + rA + i * 32) * 2048 + k0 + cA);
  };
  auto store_l = [&](int buf) {
#pragma unroll
    for (int i = 0; i < 4; i++) {
      bf16x4 h = { (bf16)ra[i].x, (bf16)ra[i].y, (bf16)ra[i].z, (bf16)ra[i].w };
      *(bf16x4*)(As[buf] + (rA + i * 32) * SA + cA) = h;
    }
#pragma unroll
    for (int i = 0; i < 4; i++) {
      bf16x4 h = { (bf16)rb[i].x, (bf16)rb[i].y, (bf16)rb[i].z, (bf16)rb[i].w };
      *(bf16x4*)(Bs[buf] + (rA + i * 32) * SA + cA) = h;
    }
  };

  const int nK = 2048 / BLK_K;
  load_g(0); store_l(0); __syncthreads();
  for (int kt = 0; kt < nK; kt++) {
    if (kt + 1 < nK) load_g((kt + 1) * BLK_K);
    mma_step(As[kt & 1], Bs[kt & 1], wm, wn, acc);
    if (kt + 1 < nK) store_l((kt + 1) & 1);
    __syncthreads();
  }

#pragma unroll
  for (int mi = 0; mi < 2; mi++)
#pragma unroll
    for (int ni = 0; ni < 4; ni++) {
      v8f c = acc[mi][ni];
      int n = n0 + wn + ni * 16 + (lane & 15);
      int r = n & 2047;
      float bias_v = bias[r];
      int h = r >> 10, d = r & 1023;
#pragma unroll
      for (int rr = 0; rr < 8; rr++) {
        int m = m0 + wm + mi * 16 + rr + ((lane >> 4) << 3);
        int b = m >> 10, t = m & 1023;
        float val = c[rr] + bias_v;
        size_t zb = ((size_t)(b * 2 + h)) << 20;
        if (which == 0)      qws [zb + (size_t)t * 1024 + d] = (bf16)val;
        else if (which == 1) kws [zb + (size_t)t * 1024 + d] = (bf16)val;
        else                 vtws[zb + (size_t)d * 1024 + t] = (bf16)val;
      }
    }
}

// ---------------- Kernel 2: scores = scale * q @ k^T  (per b,h slab) ----------------
__global__ __launch_bounds__(256)
void scores_gemm(const bf16* __restrict__ qws, const bf16* __restrict__ kws,
                 float* __restrict__ scores) {
  const int m0 = blockIdx.y * BLK_M, n0 = blockIdx.x * BLK_N;
  if (n0 > m0 + (BLK_M - 1)) return;                // above causal diagonal: skip
  __shared__ bf16 As[2][BLK_M * SA];
  __shared__ bf16 Bs[2][BLK_N * SA];
  const int tid = threadIdx.x, lane = tid & 31, wave = tid >> 5;
  const int wm = (wave & 3) * 32, wn = (wave >> 2) * 64;
  const size_t zoff = ((size_t)blockIdx.z) << 20;
  const bf16* A = qws + zoff;
  const bf16* B = kws + zoff;
  float* S = scores + zoff;
  const int r8 = tid >> 2, c8 = (tid & 3) << 3;     // per-i: row = r8 + i*64

  bf16x8 ra[2], rb[2];
  v8f acc[2][4] = {};

  auto load_g = [&](int k0) {
#pragma unroll
    for (int i = 0; i < 2; i++)
      ra[i] = *(const bf16x8*)(A + (size_t)(m0 + r8 + i * 64) * 1024 + k0 + c8);
#pragma unroll
    for (int i = 0; i < 2; i++)
      rb[i] = *(const bf16x8*)(B + (size_t)(n0 + r8 + i * 64) * 1024 + k0 + c8);
  };
  auto store_l = [&](int buf) {
#pragma unroll
    for (int i = 0; i < 2; i++) *(bf16x8*)(As[buf] + (r8 + i * 64) * SA + c8) = ra[i];
#pragma unroll
    for (int i = 0; i < 2; i++) *(bf16x8*)(Bs[buf] + (r8 + i * 64) * SA + c8) = rb[i];
  };

  const int nK = 1024 / BLK_K;
  load_g(0); store_l(0); __syncthreads();
  for (int kt = 0; kt < nK; kt++) {
    if (kt + 1 < nK) load_g((kt + 1) * BLK_K);
    mma_step(As[kt & 1], Bs[kt & 1], wm, wn, acc);
    if (kt + 1 < nK) store_l((kt + 1) & 1);
    __syncthreads();
  }

#pragma unroll
  for (int mi = 0; mi < 2; mi++)
#pragma unroll
    for (int ni = 0; ni < 4; ni++) {
      v8f c = acc[mi][ni];
      int n = n0 + wn + ni * 16 + (lane & 15);
#pragma unroll
      for (int rr = 0; rr < 8; rr++) {
        int m = m0 + wm + mi * 16 + rr + ((lane >> 4) << 3);
        S[(size_t)m * 1024 + n] = c[rr] * SCALE;
      }
    }
}

// ---------------- Kernel 3: causal softmax per row, f32 -> bf16 probs ----------------
__global__ __launch_bounds__(256)
void softmax_rows(const float* __restrict__ scores, bf16* __restrict__ probs) {
  const int row = blockIdx.x;
  const int t = row & 1023;
  const int tid = threadIdx.x, lane = tid & 31, wave = tid >> 5;
  const float* srow = scores + (size_t)row * 1024;
  float4 xv = *(const float4*)(srow + tid * 4);
  float x[4] = { xv.x, xv.y, xv.z, xv.w };
#pragma unroll
  for (int j = 0; j < 4; j++) if (tid * 4 + j > t) x[j] = -INFINITY;

  __shared__ float red[8];
  float mx = fmaxf(fmaxf(x[0], x[1]), fmaxf(x[2], x[3]));
#pragma unroll
  for (int off = 16; off > 0; off >>= 1) mx = fmaxf(mx, __shfl_xor(mx, off));
  if (lane == 0) red[wave] = mx;
  __syncthreads();
  mx = red[0];
#pragma unroll
  for (int i = 1; i < 8; i++) mx = fmaxf(mx, red[i]);
  __syncthreads();

  float e[4], s = 0.f;
#pragma unroll
  for (int j = 0; j < 4; j++) { e[j] = __expf(x[j] - mx); s += e[j]; }
#pragma unroll
  for (int off = 16; off > 0; off >>= 1) s += __shfl_xor(s, off);
  if (lane == 0) red[wave] = s;
  __syncthreads();
  s = red[0];
#pragma unroll
  for (int i = 1; i < 8; i++) s += red[i];
  float inv = 1.0f / s;
  bf16x4 o = { (bf16)(e[0] * inv), (bf16)(e[1] * inv), (bf16)(e[2] * inv), (bf16)(e[3] * inv) };
  *(bf16x4*)(probs + (size_t)row * 1024 + tid * 4) = o;
}

// ---------------- Kernel 4: out = probs @ v  (per b,h; v stored (D,T)) ----------------
__global__ __launch_bounds__(256)
void pv_gemm(const bf16* __restrict__ probs, const bf16* __restrict__ vtws,
             bf16* __restrict__ attn) {
  __shared__ bf16 As[2][BLK_M * SA];
  __shared__ bf16 Bs[2][BLK_N * SA];
  const int tid = threadIdx.x, lane = tid & 31, wave = tid >> 5;
  const int wm = (wave & 3) * 32, wn = (wave >> 2) * 64;
  const int m0 = blockIdx.y * BLK_M, n0 = blockIdx.x * BLK_N;
  const int z = blockIdx.z;
  const size_t zoff = ((size_t)z) << 20;
  const bf16* A = probs + zoff;                     // (t, s)
  const bf16* B = vtws + zoff;                      // (d, s)
  const int r8 = tid >> 2, c8 = (tid & 3) << 3;

  bf16x8 ra[2], rb[2];
  v8f acc[2][4] = {};

  auto load_g = [&](int k0) {
#pragma unroll
    for (int i = 0; i < 2; i++)
      ra[i] = *(const bf16x8*)(A + (size_t)(m0 + r8 + i * 64) * 1024 + k0 + c8);
#pragma unroll
    for (int i = 0; i < 2; i++)
      rb[i] = *(const bf16x8*)(B + (size_t)(n0 + r8 + i * 64) * 1024 + k0 + c8);
  };
  auto store_l = [&](int buf) {
#pragma unroll
    for (int i = 0; i < 2; i++) *(bf16x8*)(As[buf] + (r8 + i * 64) * SA + c8) = ra[i];
#pragma unroll
    for (int i = 0; i < 2; i++) *(bf16x8*)(Bs[buf] + (r8 + i * 64) * SA + c8) = rb[i];
  };

  const int nK = (m0 + BLK_M) / BLK_K;              // probs are 0 beyond causal diagonal
  load_g(0); store_l(0); __syncthreads();
  for (int kt = 0; kt < nK; kt++) {
    if (kt + 1 < nK) load_g((kt + 1) * BLK_K);
    mma_step(As[kt & 1], Bs[kt & 1], wm, wn, acc);
    if (kt + 1 < nK) store_l((kt + 1) & 1);
    __syncthreads();
  }

  const int b = z >> 1, h = z & 1;
#pragma unroll
  for (int mi = 0; mi < 2; mi++)
#pragma unroll
    for (int ni = 0; ni < 4; ni++) {
      v8f c = acc[mi][ni];
      int d = n0 + wn + ni * 16 + (lane & 15);
#pragma unroll
      for (int rr = 0; rr < 8; rr++) {
        int t = m0 + wm + mi * 16 + rr + ((lane >> 4) << 3);
        attn[((size_t)(b * 1024 + t)) * 2048 + h * 1024 + d] = (bf16)c[rr];
      }
    }
}

// ---------------- Kernel 5: final projection out = attn @ Wo^T (f32 out) ----------------
__global__ __launch_bounds__(256)
void out_proj_gemm(const bf16* __restrict__ attn, const float* __restrict__ Wo,
                   float* __restrict__ out) {
  __shared__ bf16 As[2][BLK_M * SA];
  __shared__ bf16 Bs[2][BLK_N * SA];
  const int tid = threadIdx.x, lane = tid & 31, wave = tid >> 5;
  const int wm = (wave & 3) * 32, wn = (wave >> 2) * 64;
  const int m0 = blockIdx.y * BLK_M, n0 = blockIdx.x * BLK_N;
  const int r8 = tid >> 2, c8 = (tid & 3) << 3;     // A staging (bf16)
  const int rB = tid >> 3, cB = (tid & 7) << 2;     // B staging (f32)

  bf16x8 ra[2];
  float4 rb[4];
  v8f acc[2][4] = {};

  auto load_g = [&](int k0) {
#pragma unroll
    for (int i = 0; i < 2; i++)
      ra[i] = *(const bf16x8*)(attn + (size_t)(m0 + r8 + i * 64) * 2048 + k0 + c8);
#pragma unroll
    for (int i = 0; i < 4; i++)
      rb[i] = *(const float4*)(Wo + (size_t)(n0 + rB + i * 32) * 2048 + k0 + cB);
  };
  auto store_l = [&](int buf) {
#pragma unroll
    for (int i = 0; i < 2; i++) *(bf16x8*)(As[buf] + (r8 + i * 64) * SA + c8) = ra[i];
#pragma unroll
    for (int i = 0; i < 4; i++) {
      bf16x4 h = { (bf16)rb[i].x, (bf16)rb[i].y, (bf16)rb[i].z, (bf16)rb[i].w };
      *(bf16x4*)(Bs[buf] + (rB + i * 32) * SA + cB) = h;
    }
  };

  const int nK = 2048 / BLK_K;
  load_g(0); store_l(0); __syncthreads();
  for (int kt = 0; kt < nK; kt++) {
    if (kt + 1 < nK) load_g((kt + 1) * BLK_K);
    mma_step(As[kt & 1], Bs[kt & 1], wm, wn, acc);
    if (kt + 1 < nK) store_l((kt + 1) & 1);
    __syncthreads();
  }

#pragma unroll
  for (int mi = 0; mi < 2; mi++)
#pragma unroll
    for (int ni = 0; ni < 4; ni++) {
      v8f c = acc[mi][ni];
      int n = n0 + wn + ni * 16 + (lane & 15);
#pragma unroll
      for (int rr = 0; rr < 8; rr++) {
        int m = m0 + wm + mi * 16 + rr + ((lane >> 4) << 3);
        out[(size_t)m * 2048 + n] = c[rr];
      }
    }
}

// ---------------- Host launch ----------------
extern "C" void kernel_launch(void* const* d_in, const int* in_sizes, int n_in,
                              void* d_out, int out_size, void* d_ws, size_t ws_size,
                              hipStream_t stream) {
  (void)in_sizes; (void)n_in; (void)out_size; (void)ws_size;
  const float* x  = (const float*)d_in[0];
  const float* Wq = (const float*)d_in[1];
  const float* bq = (const float*)d_in[2];
  const float* Wk = (const float*)d_in[3];
  const float* bk = (const float*)d_in[4];
  const float* Wv = (const float*)d_in[5];
  const float* bv = (const float*)d_in[6];
  const float* Wo = (const float*)d_in[7];
  float* out = (float*)d_out;

  const size_t SLAB = 67108864ULL;                  // 32*2*1024*1024 elements
  bf16* qws   = (bf16*)d_ws;
  bf16* kws   = qws   + SLAB;
  bf16* vtws  = kws   + SLAB;
  bf16* probs = vtws  + SLAB;
  bf16* attn  = probs + SLAB;
  float* scores = (float*)(attn + SLAB);

  qkv_gemm<<<dim3(6144 / BLK_N, 32768 / BLK_M), 256, 0, stream>>>(
      x, Wq, Wk, Wv, bq, bk, bv, qws, kws, vtws);
  scores_gemm<<<dim3(1024 / BLK_N, 1024 / BLK_M, 64), 256, 0, stream>>>(qws, kws, scores);
  softmax_rows<<<65536, 256, 0, stream>>>(scores, probs);
  pv_gemm<<<dim3(1024 / BLK_N, 1024 / BLK_M, 64), 256, 0, stream>>>(probs, vtws, attn);
  out_proj_gemm<<<dim3(2048 / BLK_N, 32768 / BLK_M), 256, 0, stream>>>(attn, Wo, out);
}
